// CircleLoss_11836929868464
// MI455X (gfx1250) — matrix operand order repaired and checked by reading, not compiled, not run on hardware
//
#include <hip/hip_runtime.h>
#include <hip/hip_bf16.h>

// ---------------------------------------------------------------------------
// CircleLoss forward for MI455X (gfx1250, wave32, WMMA).
//
// sim = normalize(x) @ normalize(x)^T  via split-f16 WMMA:
//   x = hi + lo (f16 hi, f16 residual lo);  sim ~= hi*hi + hi*lo + lo*hi
// fused with masked online logsumexp rows (pos/neg), softplus, mean.
// Workspace: 16MB hi + 16MB lo + 2*8192 floats (row loss / row valid).
// ---------------------------------------------------------------------------

typedef __attribute__((ext_vector_type(16))) _Float16 v16h;
typedef __attribute__((ext_vector_type(8)))  _Float16 h8;
typedef __attribute__((ext_vector_type(8)))  float    v8f;

#define BB 8192
#define DD 1024
#define NCOLTILE (BB / 16)     // 512 column tiles of 16
#define ROWS_PER_WG 32         // 2 row-tiles of 16 per wave
#define NWAVES 8               // 256 threads

__device__ __constant__ float kMARGIN  = 0.25f;
__device__ __constant__ float kSCALE   = 64.0f;
#define C_OP      (1.0f + 0.25f)   // 1 + margin
#define C_ON      (-0.25f)         // -margin
#define C_DELTA_P (1.0f - 0.25f)
#define C_DELTA_N (0.25f)
#define C_NEGINF  (-1e30f)

#define WMMA_F16F32(a, b, c) \
  __builtin_amdgcn_wmma_f32_16x16x32_f16(false, (a), false, (b), (short)0, (c), false, false)

// A-fragment (16x32, 16-bit): lane m holds K = kA+{0..15},
// kA = (lane>=16)*16.  p points at row*DD + k0 + kA.
__device__ __forceinline__ v16h load_a_frag(const _Float16* p) {
  union { v16h v; h8 h[2]; } u;
  u.h[0] = *(const h8*)(p);
  u.h[1] = *(const h8*)(p + 8);
  return u.v;
}

// B-fragment (32x16, 16-bit): lane n holds 16 contiguous K starting at
// (lane>=16)*16.  p points at col_row*DD + k0 + (lane>=16)*16 (32B aligned).
__device__ __forceinline__ v16h load_b_frag(const _Float16* p) {
  return *(const v16h*)(p);
}

// Masked online logsumexp update (branchless; safe when nothing seen yet).
__device__ __forceinline__ void lse_update(float& m, float& s, float x, bool active) {
  float xm = active ? x : C_NEGINF;
  float nm = fmaxf(m, xm);
  s = s * __expf(m - nm) + (active ? __expf(x - nm) : 0.0f);
  m = nm;
}

__device__ __forceinline__ void lse_merge(float& m, float& s, float om, float os) {
  float nm = fmaxf(m, om);
  s = s * __expf(m - nm) + os * __expf(om - nm);
  m = nm;
}

__device__ __forceinline__ void tile_epilogue(const v8f& c, const int* trow, int tj,
                                              int iBase, int col,
                                              float* mp, float* sp, float* mn, float* sn) {
#pragma unroll
  for (int v = 0; v < 8; ++v) {
    float s = c[v];
    bool same = (trow[v] == tj);
    bool diag = ((iBase + v) == col);
    float lp = -fmaxf(C_OP - s, 0.0f) * (s - C_DELTA_P) * kSCALE;
    float ln =  fmaxf(s - C_ON, 0.0f) * (s - C_DELTA_N) * kSCALE;
    lse_update(mp[v], sp[v], lp, same && !diag);
    lse_update(mn[v], sn[v], ln, !same);
  }
}

// ---------------------------------------------------------------------------
// Kernel 1: L2-normalize rows, emit f16 hi + f16 residual lo planes.
// ---------------------------------------------------------------------------
__global__ __launch_bounds__(256) void normalize_split(const float* __restrict__ x,
                                                       _Float16* __restrict__ hi,
                                                       _Float16* __restrict__ lo) {
  const int row = blockIdx.x;
  const float* xr = x + (size_t)row * DD;
  float ss = 0.0f;
  for (int c = threadIdx.x; c < DD; c += 256) {
    float v = xr[c];
    ss = fmaf(v, v, ss);
  }
#pragma unroll
  for (int off = 16; off > 0; off >>= 1) ss += __shfl_xor(ss, off);
  __shared__ float red[8];
  if ((threadIdx.x & 31) == 0) red[threadIdx.x >> 5] = ss;
  __syncthreads();
  float tot = 0.0f;
#pragma unroll
  for (int w = 0; w < 8; ++w) tot += red[w];
  const float inv = 1.0f / fmaxf(sqrtf(tot), 1e-12f);
  for (int c = threadIdx.x; c < DD; c += 256) {
    float v = xr[c] * inv;
    _Float16 h = (_Float16)v;
    hi[(size_t)row * DD + c] = h;
    lo[(size_t)row * DD + c] = (_Float16)(v - (float)h);
  }
}

// ---------------------------------------------------------------------------
// Kernel 2: fused split-f16 WMMA GEMM + masked online LSE + softplus per row.
// ---------------------------------------------------------------------------
__global__ __launch_bounds__(256) void circle_main(const _Float16* __restrict__ xhi,
                                                   const _Float16* __restrict__ xlo,
                                                   const int* __restrict__ targets,
                                                   float* __restrict__ row_loss,
                                                   float* __restrict__ row_cnt) {
  const int lane = threadIdx.x & 31;
  const int wave = threadIdx.x >> 5;
  const int i0 = blockIdx.x * ROWS_PER_WG;
  const int l15 = lane & 15;
  const int hiHalf = lane >> 4;
  const int kA = hiHalf * 16;    // A-fragment K offset for upper half-wave
  const int h8o = hiHalf * 8;    // C-fragment row offset for upper half-wave

  const size_t rA0 = (size_t)(i0 + l15) * DD;
  const size_t rA1 = (size_t)(i0 + 16 + l15) * DD;
  const _Float16* a0h = xhi + rA0 + kA;
  const _Float16* a0l = xlo + rA0 + kA;
  const _Float16* a1h = xhi + rA1 + kA;
  const _Float16* a1l = xlo + rA1 + kA;

  int t0[8], t1[8];
#pragma unroll
  for (int v = 0; v < 8; ++v) {
    t0[v] = targets[i0 + v + h8o];
    t1[v] = targets[i0 + 16 + v + h8o];
  }

  float mp0[8], sp0[8], mn0[8], sn0[8];
  float mp1[8], sp1[8], mn1[8], sn1[8];
#pragma unroll
  for (int v = 0; v < 8; ++v) {
    mp0[v] = C_NEGINF; sp0[v] = 0.0f; mn0[v] = C_NEGINF; sn0[v] = 0.0f;
    mp1[v] = C_NEGINF; sp1[v] = 0.0f; mn1[v] = C_NEGINF; sn1[v] = 0.0f;
  }

  for (int jt = wave; jt < NCOLTILE; jt += NWAVES) {
    const int j0 = jt * 16;
    const int col = j0 + l15;
    const size_t rB = (size_t)col * DD + (size_t)hiHalf * 16;
    const _Float16* bh = xhi + rB;
    const _Float16* bl = xlo + rB;
    const int tj = targets[col];

    v8f c0 = {};
    v8f c1 = {};
    for (int k0 = 0; k0 < DD; k0 += 32) {
      v16h A0h = load_a_frag(a0h + k0);
      v16h A0l = load_a_frag(a0l + k0);
      v16h A1h = load_a_frag(a1h + k0);
      v16h A1l = load_a_frag(a1l + k0);
      v16h Bh  = load_b_frag(bh + k0);
      v16h Bl  = load_b_frag(bl + k0);
      // hi*hi (dominant) + hi*lo + lo*hi  -> ~fp32-accurate sim
      c0 = WMMA_F16F32(A0h, Bh, c0);
      c1 = WMMA_F16F32(A1h, Bh, c1);
      c0 = WMMA_F16F32(A0h, Bl, c0);
      c1 = WMMA_F16F32(A1h, Bl, c1);
      c0 = WMMA_F16F32(A0l, Bh, c0);
      c1 = WMMA_F16F32(A1l, Bh, c1);
    }
    tile_epilogue(c0, t0, tj, i0 + h8o, col, mp0, sp0, mn0, sn0);
    tile_epilogue(c1, t1, tj, i0 + 16 + h8o, col, mp1, sp1, mn1, sn1);
  }

  // Reduce the 16 lanes sharing each row (xor masks 1..8 stay in half-wave).
#pragma unroll
  for (int off = 1; off < 16; off <<= 1) {
#pragma unroll
    for (int v = 0; v < 8; ++v) {
      lse_merge(mp0[v], sp0[v], __shfl_xor(mp0[v], off), __shfl_xor(sp0[v], off));
      lse_merge(mn0[v], sn0[v], __shfl_xor(mn0[v], off), __shfl_xor(sn0[v], off));
      lse_merge(mp1[v], sp1[v], __shfl_xor(mp1[v], off), __shfl_xor(sp1[v], off));
      lse_merge(mn1[v], sn1[v], __shfl_xor(mn1[v], off), __shfl_xor(sn1[v], off));
    }
  }

  __shared__ float smp[NWAVES][ROWS_PER_WG];
  __shared__ float ssp[NWAVES][ROWS_PER_WG];
  __shared__ float smn[NWAVES][ROWS_PER_WG];
  __shared__ float ssn[NWAVES][ROWS_PER_WG];
  if (l15 == 0) {
#pragma unroll
    for (int v = 0; v < 8; ++v) {
      smp[wave][v + h8o] = mp0[v];      ssp[wave][v + h8o] = sp0[v];
      smn[wave][v + h8o] = mn0[v];      ssn[wave][v + h8o] = sn0[v];
      smp[wave][16 + v + h8o] = mp1[v]; ssp[wave][16 + v + h8o] = sp1[v];
      smn[wave][16 + v + h8o] = mn1[v]; ssn[wave][16 + v + h8o] = sn1[v];
    }
  }
  __syncthreads();

  if (threadIdx.x < ROWS_PER_WG) {
    const int r = threadIdx.x;
    float mP = C_NEGINF, sP = 0.0f, mN = C_NEGINF, sN = 0.0f;
#pragma unroll
    for (int w = 0; w < NWAVES; ++w) {
      lse_merge(mP, sP, smp[w][r], ssp[w][r]);
      lse_merge(mN, sN, smn[w][r], ssn[w][r]);
    }
    const bool valid = (sP > 0.0f) && (sN > 0.0f);
    const float z = (mP + __logf(sP)) + (mN + __logf(sN));  // lse_p + lse_n
    const float loss = fmaxf(z, 0.0f) + log1pf(__expf(-fabsf(z)));  // softplus
    const int gr = i0 + r;
    row_loss[gr] = valid ? loss : 0.0f;
    row_cnt[gr]  = valid ? 1.0f : 0.0f;
  }
}

// ---------------------------------------------------------------------------
// Kernel 3: deterministic final reduction -> mean scalar.
// ---------------------------------------------------------------------------
__global__ __launch_bounds__(256) void circle_finalize(const float* __restrict__ row_loss,
                                                       const float* __restrict__ row_cnt,
                                                       float* __restrict__ out) {
  __shared__ float sl[256];
  __shared__ float sc[256];
  float l = 0.0f, c = 0.0f;
  for (int i = threadIdx.x; i < BB; i += 256) {
    l += row_loss[i];
    c += row_cnt[i];
  }
  sl[threadIdx.x] = l;
  sc[threadIdx.x] = c;
  __syncthreads();
  for (int off = 128; off > 0; off >>= 1) {
    if (threadIdx.x < off) {
      sl[threadIdx.x] += sl[threadIdx.x + off];
      sc[threadIdx.x] += sc[threadIdx.x + off];
    }
    __syncthreads();
  }
  if (threadIdx.x == 0) out[0] = sl[0] / fmaxf(sc[0], 1.0f);
}

extern "C" void kernel_launch(void* const* d_in, const int* in_sizes, int n_in,
                              void* d_out, int out_size, void* d_ws, size_t ws_size,
                              hipStream_t stream) {
  (void)in_sizes; (void)n_in; (void)out_size; (void)ws_size;
  const float* x       = (const float*)d_in[0];
  const int*   targets = (const int*)d_in[1];
  float*       out     = (float*)d_out;

  char* ws = (char*)d_ws;
  _Float16* hi = (_Float16*)ws;                                   // 16 MB
  _Float16* lo = hi + (size_t)BB * DD;                            // 16 MB
  float* row_loss = (float*)(ws + 2ull * BB * DD * sizeof(_Float16));
  float* row_cnt  = row_loss + BB;                                // 2*32 KB

  normalize_split<<<BB, 256, 0, stream>>>(x, hi, lo);
  circle_main<<<BB / ROWS_PER_WG, 256, 0, stream>>>(hi, lo, targets, row_loss, row_cnt);
  circle_finalize<<<1, 256, 0, stream>>>(row_loss, row_cnt, out);
}
